// TransformerBlock_39127152066581
// MI455X (gfx1250) — matrix operand order repaired and checked by reading, not compiled, not run on hardware
//
#include <hip/hip_runtime.h>
#include <math.h>

// ---------------------------------------------------------------------------
// Types for CDNA5 WMMA (wave32): V_WMMA_F32_16X16X32_BF16
// ---------------------------------------------------------------------------
typedef __bf16 v16bf __attribute__((ext_vector_type(16)));
typedef __bf16 v8bf  __attribute__((ext_vector_type(8)));
typedef float  v8f   __attribute__((ext_vector_type(8)));
typedef unsigned int u32x4 __attribute__((ext_vector_type(4)));
typedef int          i32x4 __attribute__((ext_vector_type(4)));
typedef int          i32x8 __attribute__((ext_vector_type(8)));

#if __has_builtin(__builtin_amdgcn_tensor_load_to_lds)
#define USE_TDM 1
#else
#define USE_TDM 0
#endif

static __device__ __forceinline__ v16bf cat8(v8bf a, v8bf b) {
    return __builtin_shufflevector(a, b, 0, 1, 2, 3, 4, 5, 6, 7,
                                         8, 9, 10, 11, 12, 13, 14, 15);
}

static __device__ __forceinline__ __bf16 f2bf(float f) {
    union { float f; unsigned u; } in;
    in.f = f;
    unsigned u = in.u;
    u += 0x7FFFu + ((u >> 16) & 1u);   // round-to-nearest-even
    union { unsigned short s; __bf16 b; } out;
    out.s = (unsigned short)(u >> 16);
    return out.b;
}

static __device__ __forceinline__ v8f wmma_bf16(v16bf a, v16bf b, v8f c) {
    return __builtin_amdgcn_wmma_f32_16x16x32_bf16(
        false, a, false, b, (short)0, c, false, false);
}

// DPP cross-lane exchange (pure VALU, row = 16 lanes on CDNA5 wave32)
template <int CTRL>
static __device__ __forceinline__ float dppf(float x) {
    union { float f; int i; } a, b;
    a.f = x;
    b.i = __builtin_amdgcn_update_dpp(0, a.i, CTRL, 0xF, 0xF, true);
    return b.f;
}
// reduce within each 16-lane half: quad xor1, quad xor2, half-mirror, mirror
static __device__ __forceinline__ float red16_max(float x) {
    x = fmaxf(x, dppf<0xB1>(x));
    x = fmaxf(x, dppf<0x4E>(x));
    x = fmaxf(x, dppf<0x141>(x));
    x = fmaxf(x, dppf<0x140>(x));
    return x;
}
static __device__ __forceinline__ float red16_sum(float x) {
    x += dppf<0xB1>(x);
    x += dppf<0x4E>(x);
    x += dppf<0x141>(x);
    x += dppf<0x140>(x);
    return x;
}
// swap value with the opposite 16-lane half (identity lane select)
static __device__ __forceinline__ float permx16f(float x) {
    union { float f; int i; } a, b;
    a.f = x;
    b.i = __builtin_amdgcn_permlanex16(a.i, a.i, 0x76543210u, 0xFEDCBA98u,
                                       false, false);
    return b.f;
}

// Problem constants
#define S_LEN   2048
#define D_MODEL 1024
#define N_HEADS 16
#define HEAD_DIM 64
#define HIDDEN  4096
#define BS_ROWS 4096   // B * S

// ---------------------------------------------------------------------------
// fp32 -> bf16 weight conversion WITH transpose: src[K][N] -> dst[N][K]
// ---------------------------------------------------------------------------
__global__ void cvt_t_bf16_kernel(const float* __restrict__ src,
                                  __bf16* __restrict__ dst, int K, int N) {
    int total = K * N;
    for (int i = blockIdx.x * blockDim.x + threadIdx.x; i < total;
         i += gridDim.x * blockDim.x) {
        int k = i / N, n = i - k * N;
        dst[(size_t)n * K + k] = f2bf(src[i]);
    }
}

// ---------------------------------------------------------------------------
// RMSNorm: one block (256 threads) per row of D_MODEL, bf16 output.
// Wave reduction: 4 DPP steps (within 16 lanes) + permlanex16 across halves.
// ---------------------------------------------------------------------------
__global__ void rmsnorm_kernel(const float* __restrict__ x,
                               const float* __restrict__ w,
                               __bf16* __restrict__ out) {
    const int row = blockIdx.x;
    const int tid = threadIdx.x;
    const float* xr = x + (size_t)row * D_MODEL;

    float ss = 0.f;
    for (int i = tid; i < D_MODEL; i += 256) {
        float v = xr[i];
        ss += v * v;
    }
    ss = red16_sum(ss);
    ss += permx16f(ss);          // full wave32 sum
    __shared__ float red[8];
    if ((tid & 31) == 0) red[tid >> 5] = ss;
    __syncthreads();
    float tot = red[0] + red[1] + red[2] + red[3] +
                red[4] + red[5] + red[6] + red[7];
    float inv = rsqrtf(tot / (float)D_MODEL + 1e-5f);
    for (int i = tid; i < D_MODEL; i += 256)
        out[(size_t)row * D_MODEL + i] = f2bf(xr[i] * inv * w[i]);
}

// ---------------------------------------------------------------------------
// WMMA GEMM: C[M,N] = A[M,K](bf16,row-major) * W^T  with W stored TRANSPOSED
// as Bt[N][K] (bf16). Block: 256 thr = 8 waves; tile 128(M) x 64(N).
// Per K-step the 64(n) x 32(k) B tile is DMA'd into LDS row-major [n][k] by
// the Tensor Data Mover (one tensor_load_to_lds per step, s_wait_tensorcnt),
// so B-fragments are two contiguous 16B ds loads (k = 16h + s, same order as A).
// ---------------------------------------------------------------------------
#define EPI_QK    0   // bias, bf16 out, head-split [BH][S][64]
#define EPI_VT    1   // bias, bf16 out, head-split transposed [BH][64][S]
#define EPI_RESID 2   // bias + residual, f32 out, row-major
#define EPI_SILU  3   // bias + silu, bf16 out, row-major

template <int EPI>
__global__ void gemm_wmma_kernel(const __bf16* __restrict__ A, int lda,
                                 const __bf16* __restrict__ Bt,   // [N][K]
                                 const float* __restrict__ bias,
                                 void* __restrict__ outp,
                                 const float* __restrict__ resid,
                                 int N, int K) {
    const int lane = threadIdx.x & 31;
    const int wave = threadIdx.x >> 5;           // 0..7
    const int mrow = blockIdx.x * 128 + wave * 16;
    const int nblk = blockIdx.y * 64;

    __shared__ __bf16 Bs[64 * 32];               // [n][k] row-major

    v8f acc[4] = {};

    const int arow = mrow + (lane & 15);
    const int kbase = (lane < 16) ? 0 : 16;      // A-fragment K base
    const int bn = lane & 15;
    const int bk = (lane < 16) ? 0 : 16;         // B-fragment K base
#if !USE_TDM
    const int cn = threadIdx.x >> 2;             // 0..63 : n row
    const int ck = (threadIdx.x & 3) * 8;        // 0..24 : k chunk
#endif

    for (int k0 = 0; k0 < K; k0 += 32) {
#if USE_TDM
        if (wave == 0) {
            // ---- Tensor DMA descriptor (D#): 2D tile 32(k) x 64(n) ----
            unsigned long long ga =
                (unsigned long long)(const void*)(Bt + (size_t)nblk * K + k0);
            unsigned lds = (unsigned)(unsigned long long)(void*)&Bs[0];
            u32x4 g0;
            g0.x = 1u;                                        // count=1
            g0.y = lds;                                       // lds_addr
            g0.z = (unsigned)(ga & 0xFFFFFFFFu);              // global_addr lo
            g0.w = (unsigned)((ga >> 32) & 0x01FFFFFFu) | (2u << 30); // hi|type=2
            // group1: data_size=2B; tensor_dim0=K, dim1=N; tile 32x64; stride=K
            unsigned long long q0 = 0x10000ull |
                ((unsigned long long)((unsigned)K & 0xFFFFu) << 48);
            unsigned long long q1 = ((unsigned long long)(unsigned)K >> 16) |
                ((unsigned long long)(unsigned)N << 16) |
                (32ull << 48);                                // tile_dim0 = 32
            unsigned long long q2 = 64ull |                   // tile_dim1 = 64
                ((unsigned long long)(unsigned)K << 32);      // dim0_stride lo
            unsigned long long q3 = 0ull;
            i32x8 g1;
            g1[0] = (int)(unsigned)q0; g1[1] = (int)(q0 >> 32);
            g1[2] = (int)(unsigned)q1; g1[3] = (int)(q1 >> 32);
            g1[4] = (int)(unsigned)q2; g1[5] = (int)(q2 >> 32);
            g1[6] = (int)(unsigned)q3; g1[7] = (int)(q3 >> 32);
            i32x4 gz = {0, 0, 0, 0};
#if defined(__clang_major__) && (__clang_major__ >= 23)
            i32x8 gz8 = {0, 0, 0, 0, 0, 0, 0, 0};
            __builtin_amdgcn_tensor_load_to_lds(g0, g1, gz, gz, gz8, 0);
#else
            __builtin_amdgcn_tensor_load_to_lds(g0, g1, gz, gz, 0);
#endif
            __builtin_amdgcn_s_wait_tensorcnt(0);
        }
#else
        // fallback: vectorized cooperative copy (Bt rows are k-contiguous)
        *(v8bf*)(&Bs[cn * 32 + ck]) =
            *(const v8bf*)(Bt + (size_t)(nblk + cn) * K + k0 + ck);
#endif
        __syncthreads();

        // A fragment: e -> k0+kbase+e
        const __bf16* ap = A + (size_t)arow * lda + k0 + kbase;
        v16bf afrag = cat8(*(const v8bf*)ap, *(const v8bf*)(ap + 8));
        if (k0 + 32 < K)
            __builtin_prefetch(ap + 32, 0, 0);

#pragma unroll
        for (int t = 0; t < 4; ++t) {
            const __bf16* bp = &Bs[(t * 16 + bn) * 32 + bk];
            v16bf bfrag = cat8(*(const v8bf*)bp, *(const v8bf*)(bp + 8));
            acc[t] = wmma_bf16(afrag, bfrag, acc[t]);
        }
        __syncthreads();
    }

    // epilogue: C layout -> VGPR r, lanes 0-15: M=r ; lanes 16-31: M=r+8
    const int rofs = (lane >> 4) << 3;
#pragma unroll
    for (int t = 0; t < 4; ++t) {
#pragma unroll
        for (int r = 0; r < 8; ++r) {
            int m = mrow + r + rofs;
            int n = nblk + t * 16 + (lane & 15);
            float v = acc[t][r] + bias[n];
            if constexpr (EPI == EPI_QK) {
                int b = m >> 11, s = m & (S_LEN - 1);
                int h = n >> 6, d = n & 63;
                ((__bf16*)outp)[(((size_t)(b * N_HEADS + h)) * S_LEN + s) * HEAD_DIM + d] = f2bf(v);
            } else if constexpr (EPI == EPI_VT) {
                int b = m >> 11, s = m & (S_LEN - 1);
                int h = n >> 6, d = n & 63;
                ((__bf16*)outp)[(((size_t)(b * N_HEADS + h)) * HEAD_DIM + d) * S_LEN + s] = f2bf(v);
            } else if constexpr (EPI == EPI_RESID) {
                size_t idx = (size_t)m * N + n;
                ((float*)outp)[idx] = v + resid[idx];
            } else { // EPI_SILU
                float sv = v / (1.f + __expf(-v));
                ((__bf16*)outp)[(size_t)m * N + n] = f2bf(sv);
            }
        }
    }
}

// ---------------------------------------------------------------------------
// Causal flash attention: one wave per (b,h, 16-query tile).
// Q,K : bf16 [BH][S][64] ; V : bf16 [BH][64][S] (transposed).
// Softmax row reductions are DPP-only (rows live in 16-lane halves).
// P staged through per-wave LDS, wave-local s_wait_dscnt fence.
// ---------------------------------------------------------------------------
__global__ void attention_kernel(const __bf16* __restrict__ Qb,
                                 const __bf16* __restrict__ Kb,
                                 const __bf16* __restrict__ Vt,
                                 __bf16* __restrict__ ctx) {
    const int lane = threadIdx.x & 31;
    const int wave = threadIdx.x >> 5;
    const int gw = blockIdx.x * 4 + wave;        // 0 .. BH*(S/16)-1
    const int qt = gw & 127;                     // query tile
    const int bh = gw >> 7;                      // head index
    const int b  = bh >> 4, h = bh & 15;

    const __bf16* Qp = Qb + (size_t)bh * S_LEN * HEAD_DIM;
    const __bf16* Kp = Kb + (size_t)bh * S_LEN * HEAD_DIM;
    const __bf16* Vp = Vt + (size_t)bh * HEAD_DIM * S_LEN;

    __shared__ __bf16 Ps[4][16 * 32];
    __bf16* myP = Ps[wave];

    const int kbase = (lane < 16) ? 0 : 16;      // A-fragment K base
    const int rofs = (lane >> 4) << 3;

    const int qrow = qt * 16 + (lane & 15);
    v16bf qfrag[2];
#pragma unroll
    for (int dc = 0; dc < 2; ++dc) {
        const __bf16* qp = Qp + (size_t)qrow * HEAD_DIM + dc * 32 + kbase;
        qfrag[dc] = cat8(*(const v8bf*)qp, *(const v8bf*)(qp + 8));
    }

    v8f cacc[4] = {};
    float mrow[8], lrow[8];
#pragma unroll
    for (int r = 0; r < 8; ++r) { mrow[r] = -1e30f; lrow[r] = 0.f; }

    const int nchunk = (qt * 16 + 15) / 32;
    const int bkk = (lane < 16) ? 0 : 16;        // B-fragment K base

    for (int c = 0; c <= nchunk; ++c) {
        const int key0 = c * 32;
        v8f s[2] = {};
#pragma unroll
        for (int tt = 0; tt < 2; ++tt) {
            int key = key0 + tt * 16 + (lane & 15);
#pragma unroll
            for (int dc = 0; dc < 2; ++dc) {
                const __bf16* kp = Kp + (size_t)key * HEAD_DIM + dc * 32 + bkk;
                v16bf kfrag = cat8(*(const v8bf*)kp, *(const v8bf*)(kp + 8));
                s[tt] = wmma_bf16(qfrag[dc], kfrag, s[tt]);
            }
        }
        // softmax update (rows duplicated across each 16-lane half)
#pragma unroll
        for (int r = 0; r < 8; ++r) {
            int q = qt * 16 + r + rofs;
            int kcol = lane & 15;
            float s0 = (key0 + kcol      <= q) ? s[0][r] * 0.125f : -1e30f;
            float s1 = (key0 + 16 + kcol <= q) ? s[1][r] * 0.125f : -1e30f;
            float v = red16_max(fmaxf(s0, s1));
            float mn = fmaxf(mrow[r], v);
            float sc = __expf(mrow[r] - mn);
            float p0 = __expf(s0 - mn);
            float p1 = __expf(s1 - mn);
            float ls = red16_sum(p0 + p1);
            lrow[r] = lrow[r] * sc + ls;
            mrow[r] = mn;
#pragma unroll
            for (int t = 0; t < 4; ++t) cacc[t][r] *= sc;
            myP[(r + rofs) * 32 + kcol]      = f2bf(p0);
            myP[(r + rofs) * 32 + 16 + kcol] = f2bf(p1);
        }
        // wave-local LDS fence (waves diverge -> no __syncthreads here)
        asm volatile("s_wait_dscnt 0" ::: "memory");

        const __bf16* pp = myP + (lane & 15) * 32 + kbase;
        v16bf pfrag = cat8(*(const v8bf*)pp, *(const v8bf*)(pp + 8));

        const int kk2 = key0 + bkk;
#pragma unroll
        for (int t = 0; t < 4; ++t) {
            const __bf16* vp = Vp + (size_t)(t * 16 + (lane & 15)) * S_LEN + kk2;
            v16bf vfrag = cat8(*(const v8bf*)vp, *(const v8bf*)(vp + 8));
            cacc[t] = wmma_bf16(pfrag, vfrag, cacc[t]);
        }
    }

#pragma unroll
    for (int r = 0; r < 8; ++r) {
        float inv = 1.f / lrow[r];
        int q = qt * 16 + r + rofs;
#pragma unroll
        for (int t = 0; t < 4; ++t) {
            int d = h * HEAD_DIM + t * 16 + (lane & 15);
            ctx[((size_t)(b * S_LEN + q)) * D_MODEL + d] = f2bf(cacc[t][r] * inv);
        }
    }
}

// ---------------------------------------------------------------------------
// Host launcher
// ---------------------------------------------------------------------------
extern "C" void kernel_launch(void* const* d_in, const int* in_sizes, int n_in,
                              void* d_out, int out_size, void* d_ws, size_t ws_size,
                              hipStream_t stream) {
    (void)in_sizes; (void)n_in; (void)out_size; (void)ws_size;

    const float* x       = (const float*)d_in[0];
    const float* norm1_w = (const float*)d_in[1];
    const float* norm2_w = (const float*)d_in[2];
    const float* wq = (const float*)d_in[3];  const float* bq = (const float*)d_in[4];
    const float* wk = (const float*)d_in[5];  const float* bk = (const float*)d_in[6];
    const float* wv = (const float*)d_in[7];  const float* bv = (const float*)d_in[8];
    const float* wo = (const float*)d_in[9];  const float* bo = (const float*)d_in[10];
    const float* w1 = (const float*)d_in[11]; const float* b1 = (const float*)d_in[12];
    const float* w2 = (const float*)d_in[13]; const float* b2 = (const float*)d_in[14];

    char* ws = (char*)d_ws;
    const size_t MB = 1u << 20;
    __bf16* wqb = (__bf16*)(ws + 0 * MB);    // transposed [N][K]
    __bf16* wkb = (__bf16*)(ws + 2 * MB);
    __bf16* wvb = (__bf16*)(ws + 4 * MB);
    __bf16* wob = (__bf16*)(ws + 6 * MB);
    __bf16* w1b = (__bf16*)(ws + 8 * MB);
    __bf16* w2b = (__bf16*)(ws + 16 * MB);
    __bf16* h1  = (__bf16*)(ws + 24 * MB);   // rmsnorm1(x)            [4096,1024]
    __bf16* qB  = (__bf16*)(ws + 32 * MB);   // Q  [BH][S][64]
    __bf16* kB  = (__bf16*)(ws + 40 * MB);   // K  [BH][S][64]
    __bf16* vT  = (__bf16*)(ws + 48 * MB);   // V  [BH][64][S]
    __bf16* ctxb= (__bf16*)(ws + 56 * MB);   // attention ctx          [4096,1024]
    float*  x1  = (float* )(ws + 64 * MB);   // x + attn_out (f32)     [4096,1024]
    __bf16* h2  = (__bf16*)(ws + 80 * MB);   // rmsnorm2(x1)           [4096,1024]
    __bf16* a1  = (__bf16*)(ws + 88 * MB);   // silu(h2@w1+b1)         [4096,4096]

    // 1) weights fp32 -> bf16, transposed to [N][K]
    cvt_t_bf16_kernel<<<1024, 256, 0, stream>>>(wq, wqb, D_MODEL, D_MODEL);
    cvt_t_bf16_kernel<<<1024, 256, 0, stream>>>(wk, wkb, D_MODEL, D_MODEL);
    cvt_t_bf16_kernel<<<1024, 256, 0, stream>>>(wv, wvb, D_MODEL, D_MODEL);
    cvt_t_bf16_kernel<<<1024, 256, 0, stream>>>(wo, wob, D_MODEL, D_MODEL);
    cvt_t_bf16_kernel<<<2048, 256, 0, stream>>>(w1, w1b, D_MODEL, HIDDEN);
    cvt_t_bf16_kernel<<<2048, 256, 0, stream>>>(w2, w2b, HIDDEN, D_MODEL);

    // 2) rmsnorm1
    rmsnorm_kernel<<<BS_ROWS, 256, 0, stream>>>(x, norm1_w, h1);

    // 3) QKV projections (WMMA + TDM)
    dim3 gP(BS_ROWS / 128, D_MODEL / 64);
    gemm_wmma_kernel<EPI_QK><<<gP, 256, 0, stream>>>(h1, D_MODEL, wqb, bq, qB, nullptr, D_MODEL, D_MODEL);
    gemm_wmma_kernel<EPI_QK><<<gP, 256, 0, stream>>>(h1, D_MODEL, wkb, bk, kB, nullptr, D_MODEL, D_MODEL);
    gemm_wmma_kernel<EPI_VT><<<gP, 256, 0, stream>>>(h1, D_MODEL, wvb, bv, vT, nullptr, D_MODEL, D_MODEL);

    // 4) causal attention (WMMA flash)
    attention_kernel<<<(2 * N_HEADS * (S_LEN / 16)) / 4, 128, 0, stream>>>(qB, kB, vT, ctxb);

    // 5) output projection + residual -> x1 (f32)
    gemm_wmma_kernel<EPI_RESID><<<gP, 256, 0, stream>>>(ctxb, D_MODEL, wob, bo, x1, x, D_MODEL, D_MODEL);

    // 6) rmsnorm2
    rmsnorm_kernel<<<BS_ROWS, 256, 0, stream>>>(x1, norm2_w, h2);

    // 7) FFN up + SiLU
    dim3 gU(BS_ROWS / 128, HIDDEN / 64);
    gemm_wmma_kernel<EPI_SILU><<<gU, 256, 0, stream>>>(h2, D_MODEL, w1b, b1, a1, nullptr, HIDDEN, D_MODEL);

    // 8) FFN down + residual -> d_out (f32)
    gemm_wmma_kernel<EPI_RESID><<<gP, 256, 0, stream>>>(a1, HIDDEN, w2b, b2, (float*)d_out, x1, D_MODEL, HIDDEN);
}
